// Att_SumBiGRU_40450001993985
// MI455X (gfx1250) — matrix operand order repaired and checked by reading, not compiled, not run on hardware
//
#include <hip/hip_runtime.h>
#include <hip/hip_bf16.h>

// ---------------------------------------------------------------------------
// Att_SumBiGRU for MI455X (gfx1250, wave32, WMMA).
//   V=32000, E=1024, H=1024, T=512, L=4096
// Phases (all on `stream`):
//   0) convert w_ih_f/r + w_hh_f/r to bf16; gather+convert x = emb[tokens]
//   1) WMMA bf16 GEMM: gx = x @ w_ih.T + b_ih  (4 combos, f32 out, L x 3H)
//   2) persistent recurrence: 2 teams (fwd/bwd) x 16 WGs; per step one WMMA
//      GEMV-batch (sentences A,B as rows 0,1 of the A tile), gate math on the
//      WG-local H-slice, one atomic team barrier per step.
//   3) head: Ht features -> relu GEMM -> row sums -> sigmoid.
// ---------------------------------------------------------------------------

typedef __bf16 bf16;
typedef __attribute__((ext_vector_type(16))) __bf16 v16bf;
typedef __attribute__((ext_vector_type(8)))  __bf16 v8bf;
typedef __attribute__((ext_vector_type(8)))  float  v8f;

#define Hdim 1024
#define Edim 1024
#define Ldim 4096
#define Tdim 512
#define H3   3072
#define NWG_TEAM 16
#define HS   64          // H outputs owned per workgroup (H / NWG_TEAM)

struct TeamBar { unsigned count; unsigned phase; unsigned pad[14]; };

// ---------------- phase 0: conversions ----------------
__global__ void k_cvt_w(const float* __restrict__ s0, const float* __restrict__ s1,
                        const float* __restrict__ s2, const float* __restrict__ s3,
                        bf16* __restrict__ d0, bf16* __restrict__ d1,
                        bf16* __restrict__ d2, bf16* __restrict__ d3) {
  int which = blockIdx.y;
  const float* s = which == 0 ? s0 : which == 1 ? s1 : which == 2 ? s2 : s3;
  bf16* d = which == 0 ? d0 : which == 1 ? d1 : which == 2 ? d2 : d3;
  size_t i = (size_t)blockIdx.x * blockDim.x + threadIdx.x;
  d[i] = (bf16)s[i];
}

__global__ void k_gather(const int* __restrict__ sa, const int* __restrict__ sb,
                         const float* __restrict__ emb,
                         bf16* __restrict__ xa, bf16* __restrict__ xb) {
  int which = blockIdx.y;
  const int* s = which ? sb : sa;
  bf16* x = which ? xb : xa;
  size_t i = (size_t)blockIdx.x * blockDim.x + threadIdx.x;   // [0, L*E)
  size_t t = i >> 10;
  size_t e = i & 1023;
  x[i] = (bf16)emb[(size_t)s[t] * Edim + e];
}

__global__ void k_init(const float* __restrict__ hidden,
                       bf16* __restrict__ hbf, float* __restrict__ hf32,
                       TeamBar* __restrict__ bars) {
  size_t i = (size_t)blockIdx.x * blockDim.x + threadIdx.x;   // [0, 4096)
  int team = (int)(i >> 11);
  int c    = (int)((i >> 10) & 1);
  int k    = (int)(i & 1023);
  float hv = hidden[(size_t)team * Hdim + k];                 // hidden[dir,0,:]
  hf32[((size_t)team * 2 + c) * Hdim + k] = hv;
  // parity-0 bf16 state (parity 1 is written at step 0 before it is read)
  hbf[(((size_t)team * 2 + 0) * 2 + c) * Hdim + k] = (bf16)hv;
  if (i < 2) { bars[i].count = 0u; bars[i].phase = 0u; }
}

// ---------------- phase 1: gx = x @ w_ih.T + b_ih (WMMA bf16) ----------------
__global__ __launch_bounds__(128)
void k_gemm_gx(const bf16* __restrict__ xA, const bf16* __restrict__ xB,
               const bf16* __restrict__ wf, const bf16* __restrict__ wr,
               const float* __restrict__ bif, const float* __restrict__ bir,
               float* __restrict__ gxAf, float* __restrict__ gxAr,
               float* __restrict__ gxBf, float* __restrict__ gxBr) {
  const int combo = blockIdx.y;                       // 0:Af 1:Ar 2:Bf 3:Br
  const bf16*  x    = (combo & 2) ? xB : xA;
  const bf16*  w    = (combo & 1) ? wr : wf;
  const float* bias = (combo & 1) ? bir : bif;
  float* out = (combo == 0) ? gxAf : (combo == 1) ? gxAr : (combo == 2) ? gxBf : gxBr;

  const int wave = threadIdx.x >> 5;
  const int lane = threadIdx.x & 31;
  const int hi   = lane >> 4;
  const int ln   = lane & 15;                         // A row m == B col n
  const int tile = blockIdx.x * 4 + wave;             // [0, 256*192)
  const int m0 = (tile / (H3 / 16)) * 16;
  const int n0 = (tile % (H3 / 16)) * 16;

  const bf16* arow = x + (size_t)(m0 + ln) * Edim;    // A: row-major x
  const bf16* brow = w + (size_t)(n0 + ln) * Edim;    // B col n = w_ih row n

  v8f acc = {};
  for (int kk = 0; kk < Edim; kk += 32) {
    // A 16x32 bf16: lane holds row m; halves 0..7 -> K = 16*hi + e,
    //               halves 8..15 -> K = 8 + 16*hi + e (contiguous)
    v8bf alo = *(const v8bf*)(arow + kk + hi * 16);
    v8bf ahi = *(const v8bf*)(arow + kk + 8 + hi * 16);
    v16bf a;
#pragma unroll
    for (int e = 0; e < 8; ++e) { a[e] = alo[e]; a[e + 8] = ahi[e]; }
    // B 32x16 bf16: lane holds col n; halves e -> K = 16*hi + e (contiguous)
    v16bf b = *(const v16bf*)(brow + kk + hi * 16);
    acc = __builtin_amdgcn_wmma_f32_16x16x32_bf16(false, a, false, b,
                                                  (short)0, acc, false, false);
  }
  const float bv = bias[n0 + ln];
#pragma unroll
  for (int v = 0; v < 8; ++v) {                       // C: lane=n, m = v + 8*hi
    const int mm = v + hi * 8;
    out[(size_t)(m0 + mm) * H3 + n0 + ln] = acc[v] + bv;
  }
}

// ---------------- phase 2: persistent GRU recurrence ----------------
__global__ __launch_bounds__(384)
void k_recur(const bf16* __restrict__ whhf, const bf16* __restrict__ whhr,
             const float* __restrict__ bhhf, const float* __restrict__ bhhr,
             const float* __restrict__ gxAf, const float* __restrict__ gxAr,
             const float* __restrict__ gxBf, const float* __restrict__ gxBr,
             bf16* __restrict__ hbf,      // [team][parity][chain][H]
             float* __restrict__ hf32,    // [team][chain][H]
             TeamBar* __restrict__ bars) {
  const int team = blockIdx.x >> 4;       // 0 = forward, 1 = backward
  const int wg   = blockIdx.x & (NWG_TEAM - 1);
  const int tid  = threadIdx.x;
  const int wave = tid >> 5;              // 12 waves -> 12 N-tiles of 16
  const int lane = tid & 31;
  const int hi   = lane >> 4;
  const int lm   = lane & 15;
  const int j0   = wg * HS;

  const bf16*  whh = team ? whhr : whhf;
  const float* bhh = team ? bhhr : bhhf;
  const float* gx0 = team ? gxAr : gxAf;  // chain 0 = sentence A
  const float* gx1 = team ? gxBr : gxBf;  // chain 1 = sentence B
  TeamBar* bar = bars + team;

  __shared__ float gh_s[2][3 * HS];       // [chain][local w_hh row]

  // This lane's B column = local row lr; map to global w_hh row {r,z,n}xH-slice
  const int lr   = wave * 16 + lm;
  const int grow = (lr / HS) * Hdim + j0 + (lr % HS);
  const bf16* brow = whh + (size_t)grow * Hdim;
  __builtin_prefetch(brow, 0, 3);         // global_prefetch_b8: warm L2/L0

  bf16*  hb_team = hbf  + (size_t)team * 2 * 2 * Hdim;
  float* hf_team = hf32 + (size_t)team * 2 * Hdim;

  for (int s = 0; s < Ldim; ++s) {
    const int par = s & 1;
    const bf16* hread = hb_team + (size_t)par * 2 * Hdim;   // [chain][H]
    const bf16* arow  = hread + (size_t)(lm < 2 ? lm : 0) * Hdim;

    v8f acc = {};
    for (int kk = 0; kk < Hdim; kk += 32) {
      v16bf a = {};                       // rows 2..15 of A are zero padding
      if (lm < 2) {
        v8bf alo = *(const v8bf*)(arow + kk + hi * 16);
        v8bf ahi = *(const v8bf*)(arow + kk + 8 + hi * 16);
#pragma unroll
        for (int e = 0; e < 8; ++e) { a[e] = alo[e]; a[e + 8] = ahi[e]; }
      }
      v16bf b = *(const v16bf*)(brow + kk + hi * 16);
      acc = __builtin_amdgcn_wmma_f32_16x16x32_bf16(false, a, false, b,
                                                    (short)0, acc, false, false);
    }
    if (hi == 0) {                        // C rows m=0,1 live in elems 0,1
      gh_s[0][wave * 16 + lm] = acc[0];
      gh_s[1][wave * 16 + lm] = acc[1];
    }
    __syncthreads();

    if (tid < 2 * HS) {                   // gate math for owned h-slice
      const int c  = tid >> 6;
      const int j  = tid & (HS - 1);
      const int jj = j0 + j;
      const float* gx = c ? gx1 : gx0;
      const size_t tb = (size_t)(team ? (Ldim - 1 - s) : s) * H3;
      const float ghr = gh_s[c][j]          + bhh[jj];
      const float ghz = gh_s[c][HS + j]     + bhh[Hdim + jj];
      const float ghn = gh_s[c][2 * HS + j] + bhh[2 * Hdim + jj];
      const float r = 1.f / (1.f + __expf(-(gx[tb + jj] + ghr)));
      const float z = 1.f / (1.f + __expf(-(gx[tb + Hdim + jj] + ghz)));
      const float n = tanhf(gx[tb + 2 * Hdim + jj] + r * ghn);
      const float hp = hf_team[(size_t)c * Hdim + jj];
      const float hn = (1.f - z) * n + z * hp;
      hf_team[(size_t)c * Hdim + jj] = hn;
      hb_team[(size_t)(par ^ 1) * 2 * Hdim + (size_t)c * Hdim + jj] = (bf16)hn;
    }
    __syncthreads();

    // one team-wide barrier per step (sense via monotonically rising phase)
    if (tid == 0) {
      __threadfence();
      const unsigned old = atomicAdd(&bar->count, 1u);
      if (old == NWG_TEAM - 1) {
        atomicExch(&bar->count, 0u);
        __threadfence();
        atomicAdd(&bar->phase, 1u);
      } else {
        while (__hip_atomic_load(&bar->phase, __ATOMIC_ACQUIRE,
                                 __HIP_MEMORY_SCOPE_AGENT) < (unsigned)(s + 1))
          __builtin_amdgcn_s_sleep(2);
      }
    }
    __syncthreads();
  }
}

// ---------------- phase 3: feature head ----------------
__global__ __launch_bounds__(512)
void k_head(const float* __restrict__ hf32, const float* __restrict__ W2,
            const float* __restrict__ b2, const float* __restrict__ Wl,
            const float* __restrict__ bl, float* __restrict__ out) {
  const float* hAf = hf32 + 0 * Hdim;
  const float* hBf = hf32 + 1 * Hdim;
  const float* hAb = hf32 + 2 * Hdim;
  const float* hBb = hf32 + 3 * Hdim;
  __shared__ float buf[512];
  __shared__ float red[4];
  const int j = threadIdx.x;
  float a0 = b2[j], a1 = b2[j], a2 = b2[j], a3 = b2[j];
  const float* wrow = W2 + (size_t)j * Hdim;
  for (int k = 0; k < Hdim; ++k) {
    const float w = wrow[k];
    const float af = hAf[k], bfv = hBf[k], ab = hAb[k], bb = hBb[k];
    a0 += fabsf(af - bfv) * w;
    a1 += (af * bfv) * w;
    a2 += fabsf(ab - bb) * w;
    a3 += (ab * bb) * w;
  }
  float vals[4] = {fmaxf(a0, 0.f), fmaxf(a1, 0.f), fmaxf(a2, 0.f), fmaxf(a3, 0.f)};
  for (int c = 0; c < 4; ++c) {
    buf[j] = vals[c];
    __syncthreads();
    for (int off = 256; off > 0; off >>= 1) {
      if (j < off) buf[j] += buf[j + off];
      __syncthreads();
    }
    if (j == 0) red[c] = buf[0];
    __syncthreads();
  }
  if (j == 0) {
    float s = bl[0];
    for (int c = 0; c < 4; ++c) s += red[c] * Wl[c];
    out[0] = 1.f / (1.f + __expf(-s));
  }
}

// ---------------- launcher ----------------
extern "C" void kernel_launch(void* const* d_in, const int* in_sizes, int n_in,
                              void* d_out, int out_size, void* d_ws, size_t ws_size,
                              hipStream_t stream) {
  (void)in_sizes; (void)n_in; (void)out_size; (void)ws_size;
  const int*   sentA  = (const int*)  d_in[0];
  const int*   sentB  = (const int*)  d_in[1];
  const float* hidden = (const float*)d_in[2];
  const float* emb    = (const float*)d_in[3];
  const float* w_ih_f = (const float*)d_in[4];
  const float* w_hh_f = (const float*)d_in[5];
  const float* b_ih_f = (const float*)d_in[6];
  const float* b_hh_f = (const float*)d_in[7];
  const float* w_ih_r = (const float*)d_in[8];
  const float* w_hh_r = (const float*)d_in[9];
  const float* b_ih_r = (const float*)d_in[10];
  const float* b_hh_r = (const float*)d_in[11];
  const float* W2 = (const float*)d_in[12];
  const float* b2 = (const float*)d_in[13];
  const float* Wl = (const float*)d_in[14];
  const float* bl = (const float*)d_in[15];

  char* p = (char*)d_ws;
  auto take = [&](size_t bytes) -> char* {
    char* r = p;
    p += (bytes + 255) & ~(size_t)255;
    return r;
  };
  bf16* wihf_b = (bf16*)take((size_t)H3 * Edim * sizeof(bf16));
  bf16* wihr_b = (bf16*)take((size_t)H3 * Edim * sizeof(bf16));
  bf16* whhf_b = (bf16*)take((size_t)H3 * Hdim * sizeof(bf16));
  bf16* whhr_b = (bf16*)take((size_t)H3 * Hdim * sizeof(bf16));
  bf16* xA_b   = (bf16*)take((size_t)Ldim * Edim * sizeof(bf16));
  bf16* xB_b   = (bf16*)take((size_t)Ldim * Edim * sizeof(bf16));
  float* gxAf  = (float*)take((size_t)Ldim * H3 * sizeof(float));
  float* gxAr  = (float*)take((size_t)Ldim * H3 * sizeof(float));
  float* gxBf  = (float*)take((size_t)Ldim * H3 * sizeof(float));
  float* gxBr  = (float*)take((size_t)Ldim * H3 * sizeof(float));
  bf16*  hbf   = (bf16*) take((size_t)2 * 2 * 2 * Hdim * sizeof(bf16));
  float* hf32  = (float*)take((size_t)2 * 2 * Hdim * sizeof(float));
  TeamBar* bars = (TeamBar*)take(2 * sizeof(TeamBar));

  const dim3 b256(256);
  k_cvt_w<<<dim3(H3 * Edim / 256, 4), b256, 0, stream>>>(
      w_ih_f, w_ih_r, w_hh_f, w_hh_r, wihf_b, wihr_b, whhf_b, whhr_b);
  k_gather<<<dim3(Ldim * Edim / 256, 2), b256, 0, stream>>>(
      sentA, sentB, emb, xA_b, xB_b);
  k_init<<<dim3(16), b256, 0, stream>>>(hidden, hbf, hf32, bars);
  k_gemm_gx<<<dim3((256 * (H3 / 16)) / 4, 4), dim3(128), 0, stream>>>(
      xA_b, xB_b, wihf_b, wihr_b, b_ih_f, b_ih_r, gxAf, gxAr, gxBf, gxBr);
  k_recur<<<dim3(2 * NWG_TEAM), dim3(384), 0, stream>>>(
      whhf_b, whhr_b, b_hh_f, b_hh_r, gxAf, gxAr, gxBf, gxBr, hbf, hf32, bars);
  k_head<<<dim3(1), dim3(512), 0, stream>>>(hf32, W2, b2, Wl, bl, (float*)d_out);
}
